// MultiLayerLSTM_6012954214438
// MI455X (gfx1250) — matrix operand order repaired and checked by reading, not compiled, not run on hardware
//
#include <hip/hip_runtime.h>
#include <hip/hip_bf16.h>

// ---------------------------------------------------------------------------
// MultiLayerLSTM on gfx1250 (MI455X): bf16 WMMA GEMMs + per-timestep fused
// recurrent step with async global->LDS staging of h_prev.
// B=32, T=512, D=H=1024, L=3, gate order (i,f,g,o).
// ---------------------------------------------------------------------------

typedef __attribute__((ext_vector_type(16))) __bf16 v16bf;
typedef __attribute__((ext_vector_type(8)))  __bf16 v8bf;
typedef __attribute__((ext_vector_type(8)))  float  v8f;
typedef __attribute__((__vector_size__(4 * sizeof(int)))) int v4i;

#define LSTM_B 32
#define LSTM_T 512
#define LSTM_H 1024
#define LSTM_4H 4096
#define LSTM_M (LSTM_B * LSTM_T)   // 16384 rows of x_proj

// gfx1250 async global->LDS copy path (ASYNCcnt-tracked), with fallback.
#if defined(__HIP_DEVICE_COMPILE__) &&                                        \
    __has_builtin(__builtin_amdgcn_global_load_async_to_lds_b128) &&          \
    __has_builtin(__builtin_amdgcn_s_wait_asynccnt)
#define USE_ASYNC_LDS 1
#else
#define USE_ASYNC_LDS 0
#endif

// ---------------------------------------------------------------------------
// f32 -> bf16 conversion (grid-stride)
// ---------------------------------------------------------------------------
__global__ void k_cvt_f32_bf16(const float* __restrict__ in,
                               __bf16* __restrict__ out, int n) {
  int i = blockIdx.x * blockDim.x + threadIdx.x;
  int stride = gridDim.x * blockDim.x;
  for (; i < n; i += stride) out[i] = (__bf16)in[i];
}

// ---------------------------------------------------------------------------
// x_proj = X(bf16)[16384x1024] * W(bf16)[4096x1024]^T + b_ih + b_hh  -> f32
// grid = (M/16, N/512), block = 256 (8 waves). Wave computes a 16x64 tile
// (4 accumulators), K accumulated in chunks of 32 with v_wmma_f32_16x16x32_bf16.
// ---------------------------------------------------------------------------
__global__ void __launch_bounds__(256)
k_gemm_xproj(const __bf16* __restrict__ X, const __bf16* __restrict__ W,
             const float* __restrict__ bih, const float* __restrict__ bhh,
             float* __restrict__ C) {
  const int tid  = threadIdx.x;
  const int wave = tid >> 5;
  const int lane = tid & 31;
  const int m0   = blockIdx.x * 16;
  const int n0   = blockIdx.y * 512 + wave * 64;
  const int half = lane >> 4;      // 0: lanes 0-15, 1: lanes 16-31
  const int mrow = lane & 15;      // A row handled by this lane
  const int ncol = lane & 15;      // B column handled by this lane

  v8f acc[4] = {v8f{}, v8f{}, v8f{}, v8f{}};

  const __bf16* ap = X + (size_t)(m0 + mrow) * LSTM_H + half * 8;
  for (int k = 0; k < LSTM_H; k += 32) {
    // A fragment 16x32: lane<16 K={0..7,16..23}, lane>=16 K={8..15,24..31}
    union { v16bf v; v8bf h[2]; } a;
    a.h[0] = *(const v8bf*)(ap + k);
    a.h[1] = *(const v8bf*)(ap + k + 16);
#pragma unroll
    for (int j = 0; j < 4; ++j) {
      // B fragment 32x16: column = row of W (row-major), K={0..7,16..23}/{8..15,24..31}
      const __bf16* bp = W + (size_t)(n0 + j * 16 + ncol) * LSTM_H + k + half * 8;
      union { v16bf v; v8bf h[2]; } bu;
      bu.h[0] = *(const v8bf*)bp;
      bu.h[1] = *(const v8bf*)(bp + 16);
      v16bf b = bu.v;
      acc[j] = __builtin_amdgcn_wmma_f32_16x16x32_bf16(
          false, a.v, false, b, (short)0, acc[j], false, false);
    }
  }

  // D layout: VGPR r -> (M = r + 8*half, N = lane&15)
  const int mb = half * 8;
#pragma unroll
  for (int j = 0; j < 4; ++j) {
    const int col = n0 + j * 16 + ncol;
    const float bias = bih[col] + bhh[col];
#pragma unroll
    for (int r = 0; r < 8; ++r) {
      C[(size_t)(m0 + mb + r) * LSTM_4H + col] = acc[j][r] + bias;
    }
  }
}

// ---------------------------------------------------------------------------
// One LSTM timestep. grid = 32 blocks (hidden slice of 32 units each),
// block = 256 threads = 8 waves. h_prev [32x1024] bf16 is staged into LDS
// once per block via GLOBAL_LOAD_ASYNC_TO_LDS_B128 (ASYNCcnt), then each
// wave pair computes one gate's 32x32 tile with WMMA reading A from LDS and
// B (W_hh rows) from global/L2. Gates staged through LDS, then the fused
// sigmoid/tanh cell update produces c, h (bf16 always, f32 for last layer),
// and hT/cT at t == T-1.
// Dynamic LDS: 64 KB h tile + 16 KB gate buffer = 80 KB.
// ---------------------------------------------------------------------------
__global__ void __launch_bounds__(256)
k_lstm_step(const float* __restrict__ xproj,   // [16384 x 4096] f32 (bias folded)
            const __bf16* __restrict__ Whh,    // [4096 x 1024] bf16
            __bf16* __restrict__ ybf,          // [B,T,H] bf16 (this layer's output)
            float* __restrict__ yf32,          // d_out outputs region
            float* __restrict__ cbuf,          // [B,H] f32 cell state
            float* __restrict__ hs_out,        // d_out hT slice for this layer
            float* __restrict__ cs_out,        // d_out cT slice for this layer
            int t, int write_f32, int last) {
  extern __shared__ char smem_raw[];
  __bf16* sh   = (__bf16*)smem_raw;            // [32][1024] bf16 = 64 KB
  float* gbase = (float*)(smem_raw + 32 * LSTM_H * sizeof(__bf16)); // [4][32][32]
#define GB(g, m, j) gbase[((((g) << 5) + (m)) << 5) + (j)]

  const int tid  = threadIdx.x;
  const int wave = tid >> 5;
  const int lane = tid & 31;
  const int g    = wave >> 1;                  // gate 0..3 (i,f,g,o)
  const int j0   = blockIdx.x * 32;
  const int colbase = j0 + (wave & 1) * 16;
  const int half = lane >> 4;
  const int mrow = lane & 15;
  const int ncol = lane & 15;

  // ---- stage h_prev into LDS (skip entirely at t==0: h0 == 0) -------------
  if (t > 0) {
    const int m  = tid >> 3;                   // 8 threads per batch row
    const int cb = (tid & 7) * 128;            // 128 elems = 16 x b128 each
    const __bf16* gsrc = ybf + ((size_t)m * LSTM_T + (t - 1)) * LSTM_H + cb;
    __bf16* ldst = sh + m * LSTM_H + cb;
#if USE_ASYNC_LDS
#pragma unroll
    for (int c = 0; c < 16; ++c) {
      __builtin_amdgcn_global_load_async_to_lds_b128(
          (v4i*)(gsrc + c * 8), (v4i*)(ldst + c * 8), 0, 0);
    }
    __builtin_amdgcn_s_wait_asynccnt(0);
#else
#pragma unroll
    for (int c = 0; c < 16; ++c) {
      *(v8bf*)(ldst + c * 8) = *(const v8bf*)(gsrc + c * 8);
    }
#endif
  }
  __syncthreads();

  // ---- recurrent GEMM: gates[32, colbase..+16] over K=1024 ----------------
  v8f c0 = v8f{};   // batch rows 0..15
  v8f c1 = v8f{};   // batch rows 16..31

  if (t > 0) {
    const __bf16* a0p  = sh + mrow * LSTM_H + half * 8;
    const __bf16* a1p  = sh + (mrow + 16) * LSTM_H + half * 8;
    const __bf16* wrow = Whh + (size_t)(g * LSTM_H + colbase + ncol) * LSTM_H;
    for (int k = 0; k < LSTM_H; k += 32) {
      union { v16bf v; v8bf h[2]; } a0, a1;
      a0.h[0] = *(const v8bf*)(a0p + k);
      a0.h[1] = *(const v8bf*)(a0p + k + 16);
      a1.h[0] = *(const v8bf*)(a1p + k);
      a1.h[1] = *(const v8bf*)(a1p + k + 16);
      union { v16bf v; v8bf h[2]; } bu;
      bu.h[0] = *(const v8bf*)(wrow + k + half * 8);
      bu.h[1] = *(const v8bf*)(wrow + k + 16 + half * 8);
      v16bf b = bu.v;
      c0 = __builtin_amdgcn_wmma_f32_16x16x32_bf16(false, a0.v, false, b,
                                                   (short)0, c0, false, false);
      c1 = __builtin_amdgcn_wmma_f32_16x16x32_bf16(false, a1.v, false, b,
                                                   (short)0, c1, false, false);
    }
  }

  // ---- stage gate tiles in LDS (D layout -> [gate][m][jlocal]) ------------
  const int jl = (wave & 1) * 16 + ncol;
  const int mb = half * 8;
#pragma unroll
  for (int r = 0; r < 8; ++r) {
    GB(g, mb + r, jl)      = c0[r];
    GB(g, 16 + mb + r, jl) = c1[r];
  }
  __syncthreads();

  // ---- fused cell update: 32 batches x 32 hidden units, 4/thread ----------
#pragma unroll
  for (int p = 0; p < 4; ++p) {
    const int e  = tid * 4 + p;
    const int m  = e >> 5;
    const int jlocal = e & 31;
    const int j  = j0 + jlocal;
    const size_t xr = ((size_t)m * LSTM_T + t) * LSTM_4H;
    float gi = GB(0, m, jlocal) + xproj[xr + 0 * LSTM_H + j];
    float gf = GB(1, m, jlocal) + xproj[xr + 1 * LSTM_H + j];
    float gg = GB(2, m, jlocal) + xproj[xr + 2 * LSTM_H + j];
    float go = GB(3, m, jlocal) + xproj[xr + 3 * LSTM_H + j];
    gi = 1.0f / (1.0f + __expf(-gi));
    gf = 1.0f / (1.0f + __expf(-gf));
    gg = tanhf(gg);
    go = 1.0f / (1.0f + __expf(-go));
    const float cprev = (t > 0) ? cbuf[m * LSTM_H + j] : 0.0f;
    const float cn = gf * cprev + gi * gg;
    const float h  = go * tanhf(cn);
    cbuf[m * LSTM_H + j] = cn;
    const size_t yi = ((size_t)m * LSTM_T + t) * LSTM_H + j;
    ybf[yi] = (__bf16)h;
    if (write_f32) yf32[yi] = h;
    if (last) {
      hs_out[m * LSTM_H + j] = h;
      cs_out[m * LSTM_H + j] = cn;
    }
  }
#undef GB
}

// ---------------------------------------------------------------------------
// Host-side orchestration
// ---------------------------------------------------------------------------
extern "C" void kernel_launch(void* const* d_in, const int* in_sizes, int n_in,
                              void* d_out, int out_size, void* d_ws, size_t ws_size,
                              hipStream_t stream) {
  (void)in_sizes; (void)n_in; (void)out_size; (void)ws_size;

  const float* x = (const float*)d_in[0];
  const float* Wih[3];
  const float* Whh[3];
  const float* bih[3];
  const float* bhh[3];
  for (int l = 0; l < 3; ++l) {
    Wih[l] = (const float*)d_in[1 + 4 * l];
    Whh[l] = (const float*)d_in[2 + 4 * l];
    bih[l] = (const float*)d_in[3 + 4 * l];
    bhh[l] = (const float*)d_in[4 + 4 * l];
  }

  // ---- workspace carve-up -------------------------------------------------
  char* w = (char*)d_ws;
  const size_t act_bytes = (size_t)LSTM_M * LSTM_H * sizeof(__bf16);  // 32 MB
  const size_t wt_bytes  = (size_t)LSTM_4H * LSTM_H * sizeof(__bf16); //  8 MB
  __bf16* bufX = (__bf16*)w; w += act_bytes;
  __bf16* bufY = (__bf16*)w; w += act_bytes;
  __bf16* wihbf[3];
  __bf16* whhbf[3];
  for (int l = 0; l < 3; ++l) { wihbf[l] = (__bf16*)w; w += wt_bytes; }
  for (int l = 0; l < 3; ++l) { whhbf[l] = (__bf16*)w; w += wt_bytes; }
  float* xproj = (float*)w; w += (size_t)LSTM_M * LSTM_4H * sizeof(float); // 256 MB
  float* cbuf  = (float*)w; w += (size_t)LSTM_B * LSTM_H * sizeof(float);

  // ---- d_out layout: outputs [B,T,H] | hs [L,B,H] | cs [L,B,H] ------------
  float* ydst = (float*)d_out;
  float* hs   = ydst + (size_t)LSTM_M * LSTM_H;
  float* cs   = hs + (size_t)3 * LSTM_B * LSTM_H;

  // ---- one-time f32 -> bf16 conversions -----------------------------------
  k_cvt_f32_bf16<<<4096, 256, 0, stream>>>(x, bufX, LSTM_M * LSTM_H);
  for (int l = 0; l < 3; ++l) {
    k_cvt_f32_bf16<<<2048, 256, 0, stream>>>(Wih[l], wihbf[l], LSTM_4H * LSTM_H);
    k_cvt_f32_bf16<<<2048, 256, 0, stream>>>(Whh[l], whhbf[l], LSTM_4H * LSTM_H);
  }

  // ---- layers: big input GEMM, then 512 sequential fused steps ------------
  const size_t step_lds = 32 * LSTM_H * sizeof(__bf16) + 4 * 32 * 32 * sizeof(float);
  __bf16* ping[2] = {bufX, bufY};
  for (int l = 0; l < 3; ++l) {
    __bf16* in_act  = ping[l & 1];
    __bf16* out_act = ping[(l + 1) & 1];
    k_gemm_xproj<<<dim3(LSTM_M / 16, LSTM_4H / 512), 256, 0, stream>>>(
        in_act, wihbf[l], bih[l], bhh[l], xproj);
    for (int t = 0; t < LSTM_T; ++t) {
      k_lstm_step<<<LSTM_H / 32, 256, step_lds, stream>>>(
          xproj, whhbf[l], out_act, ydst, cbuf,
          hs + (size_t)l * LSTM_B * LSTM_H,
          cs + (size_t)l * LSTM_B * LSTM_H,
          t, (l == 2) ? 1 : 0, (t == LSTM_T - 1) ? 1 : 0);
    }
  }
}
